// EnhancedStableNCA_38250978738678
// MI455X (gfx1250) — compile-verified, matrix operand order checked by reading, not run on hardware
//
#include <hip/hip_runtime.h>
#include <cmath>

#define CH   16
#define HID  128
#define NB   32
#define H_   128
#define W_   128
#define TILE 16
#define HALO 18

typedef __attribute__((ext_vector_type(16))) _Float16 v16h;
typedef __attribute__((ext_vector_type(8)))  _Float16 v8h;
typedef __attribute__((ext_vector_type(8)))  float    v8f;

// ---- LDS layout (bytes), all offsets 32B aligned ----
#define OFF_XS     0            // float[16*18*18]           20736
#define OFF_RND    20736        // float[256]                 1024
#define OFF_ALIVE  21760        // float[256]                 1024
#define OFF_B1     22784        // float[128]                  512
#define OFF_B2     23296        // float[128]                  512
#define OFF_B3     23808        // float[16]                    64
#define OFF_W1     23872        // _Float16[128*64]          16384
#define OFF_W2     40256        // _Float16[128*128]         32768
#define OFF_W3     73024        // _Float16[16*128]           4096
#define OFF_OBS    77120        // _Float16[256*64]          32768
#define OFF_H1     109888       // _Float16[256*128]         65536
#define OFF_H2     175424       // _Float16[256*128]         65536
#define SMEM_BYTES 240960

#define WMMA_F16(A, B, C) \
  __builtin_amdgcn_wmma_f32_16x16x32_f16(false, (A), false, (B), (short)0, (C), false, false)

// Load one 16x32 f16 WMMA fragment (A or B operand) from an LDS matrix whose
// rows are [row][kstride] with K contiguous. Per the CDNA5 ISA layout,
// lanes 0-15 hold row=row0+lane, K in {kt*32+0..7, kt*32+16..23};
// lanes 16-31 hold row=row0+lane-16, K in {kt*32+8..15, kt*32+24..31}.
// Each half is two contiguous 16-byte runs -> two ds_load_b128.
__device__ __forceinline__ v16h ld_frag(const _Float16* base, int row0,
                                        int kstride, int kt, int lane) {
  int r  = lane & 15;
  int hi = lane >> 4;
  const _Float16* p = base + (row0 + r) * kstride + kt * 32 + hi * 8;
  v8h lo = *(const v8h*)p;
  v8h hv = *(const v8h*)(p + 16);
  v16h o;
#pragma unroll
  for (int i = 0; i < 8; ++i) { o[i] = lo[i]; o[i + 8] = hv[i]; }
  return o;
}

// D-tile epilogue with weights-as-A orientation: per lane the 8 accumulator
// values are CONSECUTIVE hidden units n for one pixel m, so bias+ReLU+cvt
// packs into a single 16-byte ds_store_b128.
__device__ __forceinline__ void store_relu_pack8(_Float16* dst, const v8f& acc,
                                                 const v8f& bias) {
  v8h h;
#pragma unroll
  for (int r = 0; r < 8; ++r)
    h[r] = (_Float16)fmaxf(acc[r] + bias[r], 0.0f);
  *(v8h*)dst = h;
}

__global__ void __launch_bounds__(256)
nca_step(const float* __restrict__ xin, const float* __restrict__ rnd_g,
         const float* __restrict__ w1, const float* __restrict__ b1,
         const float* __restrict__ w2, const float* __restrict__ b2,
         const float* __restrict__ w3, const float* __restrict__ b3,
         float* __restrict__ xout) {
  extern __shared__ __align__(16) char smem[];
  float*    xs    = (float*)(smem + OFF_XS);
  float*    rnd   = (float*)(smem + OFF_RND);
  float*    alive = (float*)(smem + OFF_ALIVE);
  float*    b1f   = (float*)(smem + OFF_B1);
  float*    b2f   = (float*)(smem + OFF_B2);
  float*    b3f   = (float*)(smem + OFF_B3);
  _Float16* w1h   = (_Float16*)(smem + OFF_W1);
  _Float16* w2h   = (_Float16*)(smem + OFF_W2);
  _Float16* w3h   = (_Float16*)(smem + OFF_W3);
  _Float16* obs   = (_Float16*)(smem + OFF_OBS);
  _Float16* h1    = (_Float16*)(smem + OFF_H1);
  _Float16* h2    = (_Float16*)(smem + OFF_H2);

  const int tid  = threadIdx.x;
  const int lane = tid & 31;
  const int wv   = tid >> 5;
  const int bx   = blockIdx.x * TILE;
  const int by   = blockIdx.y * TILE;
  const int b    = blockIdx.z;

  const float* xb = xin  + (size_t)b * CH * H_ * W_;
  float*       ob = xout + (size_t)b * CH * H_ * W_;

  // ---- Phase A: halo tile (zero-padded), rand tile, weights (f32->f16) ----
  for (int i = tid; i < CH * HALO * HALO; i += 256) {
    int c  = i / (HALO * HALO);
    int r  = i - c * (HALO * HALO);
    int ly = r / HALO, lx = r - ly * HALO;
    int gy = by + ly - 1, gx = bx + lx - 1;
    float v = 0.0f;
    if (gy >= 0 && gy < H_ && gx >= 0 && gx < W_)
      v = xb[c * H_ * W_ + gy * W_ + gx];
    xs[i] = v;
  }
  {
    int ly = tid >> 4, lx = tid & 15;
    rnd[tid] = rnd_g[(by + ly) * W_ + bx + lx];
  }
  for (int i = tid; i < HID * 64;  i += 256) w1h[i] = (_Float16)w1[i];
  for (int i = tid; i < HID * HID; i += 256) w2h[i] = (_Float16)w2[i];
  for (int i = tid; i < CH * HID;  i += 256) w3h[i] = (_Float16)w3[i];
  if (tid < HID) { b1f[tid] = b1[tid]; b2f[tid] = b2[tid]; }
  if (tid < CH)  { b3f[tid] = b3[tid]; }
  __syncthreads();

  // ---- Phase B: stencils -> obs[pixel][64] f16, alive mask ----
  {
    const int p  = tid;
    const int ly = p >> 4, lx = p & 15;
    float amax = -1e30f;
    for (int c = 0; c < CH; ++c) {
      const float* xc = xs + c * HALO * HALO + ly * HALO + lx;
      float s00 = xc[0],        s01 = xc[1],          s02 = xc[2];
      float s10 = xc[HALO],     s11 = xc[HALO + 1],   s12 = xc[HALO + 2];
      float s20 = xc[2 * HALO], s21 = xc[2*HALO + 1], s22 = xc[2*HALO + 2];
      float gxv = (s02 - s00 + 2.0f * (s12 - s10) + s22 - s20) * 0.125f;
      float gyv = (s20 - s00 + 2.0f * (s21 - s01) + s22 - s02) * 0.125f;
      float lap = 0.05f * (s00 + s02 + s20 + s22) +
                  0.20f * (s01 + s10 + s12 + s21) - s11;
      _Float16* op = obs + p * 64;
      op[c]      = (_Float16)s11;
      op[16 + c] = (_Float16)gxv;
      op[32 + c] = (_Float16)gyv;
      op[48 + c] = (_Float16)lap;
      if (c == 3) {
        amax = fmaxf(fmaxf(fmaxf(s00, s01), fmaxf(s02, s10)),
                     fmaxf(fmaxf(s11, s12), fmaxf(s20, fmaxf(s21, s22))));
      }
    }
    alive[p] = (amax > 0.01f) ? 1.0f : 0.0f;
  }
  __syncthreads();

  const int col = lane & 15;
  const int hi  = lane >> 4;
  const int p0  = (wv * 2) * 16;       // this wave's first pixel tile (N dim)
  const int p1  = (wv * 2 + 1) * 16;   // second pixel tile

  // GEMM orientation: A = weights [n][k], B = activations [m][k] (both read
  // with ld_frag), D: lane <-> pixel m, VGPR r <-> hidden unit n. Two
  // independent accumulator chains (one per pixel tile) share each weight
  // fragment: halves weight LDS traffic and fills the WMMA RAW hazard window.

  // ---- Layer 1: h1[m][n] = relu(obs x w1^T + b1) ----
  {
    v16h b00 = ld_frag(obs, p0, 64, 0, lane);
    v16h b01 = ld_frag(obs, p0, 64, 1, lane);
    v16h b10 = ld_frag(obs, p1, 64, 0, lane);
    v16h b11 = ld_frag(obs, p1, 64, 1, lane);
    const int mpx0 = p0 + col, mpx1 = p1 + col;
    for (int mt = 0; mt < 8; ++mt) {
      v16h a0 = ld_frag(w1h, mt * 16, 64, 0, lane);
      v16h a1 = ld_frag(w1h, mt * 16, 64, 1, lane);
      v8f acc0 = {}, acc1 = {};
      acc0 = WMMA_F16(a0, b00, acc0);
      acc1 = WMMA_F16(a0, b10, acc1);
      acc0 = WMMA_F16(a1, b01, acc0);
      acc1 = WMMA_F16(a1, b11, acc1);
      const int nbase = mt * 16 + 8 * hi;
      v8f bias = *(const v8f*)(b1f + nbase);
      store_relu_pack8(h1 + mpx0 * HID + nbase, acc0, bias);
      store_relu_pack8(h1 + mpx1 * HID + nbase, acc1, bias);
    }
  }
  __syncthreads();

  // ---- Layer 2: h2[m][n] = relu(h1 x w2^T + b2) ----
  {
    v16h bf0[4], bf1[4];
#pragma unroll
    for (int k = 0; k < 4; ++k) {
      bf0[k] = ld_frag(h1, p0, HID, k, lane);
      bf1[k] = ld_frag(h1, p1, HID, k, lane);
    }
    const int mpx0 = p0 + col, mpx1 = p1 + col;
    for (int mt = 0; mt < 8; ++mt) {
      v8f acc0 = {}, acc1 = {};
#pragma unroll
      for (int k = 0; k < 4; ++k) {
        v16h aw = ld_frag(w2h, mt * 16, HID, k, lane);
        acc0 = WMMA_F16(aw, bf0[k], acc0);
        acc1 = WMMA_F16(aw, bf1[k], acc1);
      }
      const int nbase = mt * 16 + 8 * hi;
      v8f bias = *(const v8f*)(b2f + nbase);
      store_relu_pack8(h2 + mpx0 * HID + nbase, acc0, bias);
      store_relu_pack8(h2 + mpx1 * HID + nbase, acc1, bias);
    }
  }
  __syncthreads();

  // ---- Layer 3 + fused update/mask/alive/seed/tanh + global store ----
  // D: lane <-> pixel, VGPR r <-> channel ch = r + 8*hi. Global stores are
  // 16-lane contiguous along gx.
  {
    v8f acc0 = {}, acc1 = {};
#pragma unroll
    for (int k = 0; k < 4; ++k) {
      v16h aw  = ld_frag(w3h, 0, HID, k, lane);
      v16h bp0 = ld_frag(h2, p0, HID, k, lane);
      v16h bp1 = ld_frag(h2, p1, HID, k, lane);
      acc0 = WMMA_F16(aw, bp0, acc0);
      acc1 = WMMA_F16(aw, bp1, acc1);
    }
    const int chbase = 8 * hi;
    v8f bias3 = *(const v8f*)(b3f + chbase);
#pragma unroll
    for (int t = 0; t < 2; ++t) {
      const v8f& acc = t ? acc1 : acc0;
      const int m  = (t ? p1 : p0) + col;
      const int ly = m >> 4, lx = m & 15;
      const int gy = by + ly, gx = bx + lx;
      const float mk = (rnd[m] < 0.5f) ? 1.0f : 0.0f;
      const float av = alive[m];
      const bool seedpix = (gy == 32 && gx == 32);
#pragma unroll
      for (int r = 0; r < 8; ++r) {
        const int ch = chbase + r;
        float upd  = (acc[r] + bias3[r]) * 0.05f;
        float oldv = xs[ch * HALO * HALO + (ly + 1) * HALO + (lx + 1)];
        float v = (oldv + upd * mk) * av;
        if (seedpix && ch >= 3) v = 1.0f;
        ob[ch * H_ * W_ + gy * W_ + gx] = tanhf(v);
      }
    }
  }
}

extern "C" void kernel_launch(void* const* d_in, const int* in_sizes, int n_in,
                              void* d_out, int out_size, void* d_ws, size_t ws_size,
                              hipStream_t stream) {
  const float* x   = (const float*)d_in[0];
  const float* rnd = (const float*)d_in[1];
  const float* w1  = (const float*)d_in[2];
  const float* b1  = (const float*)d_in[3];
  const float* w2  = (const float*)d_in[4];
  const float* b2  = (const float*)d_in[5];
  const float* w3  = (const float*)d_in[6];
  const float* b3  = (const float*)d_in[7];
  float* out = (float*)d_out;
  float* ws  = (float*)d_ws;

  (void)hipFuncSetAttribute((const void*)nca_step,
                            hipFuncAttributeMaxDynamicSharedMemorySize,
                            SMEM_BYTES);

  dim3 grid(W_ / TILE, H_ / TILE, NB);
  dim3 blk(256);
  const float* src = x;
  for (int s = 0; s < 8; ++s) {
    float* dst = (s & 1) ? out : ws;   // step 7 (last) lands in d_out
    nca_step<<<grid, blk, SMEM_BYTES, stream>>>(
        src, rnd + (size_t)s * H_ * W_, w1, b1, w2, b2, w3, b3, dst);
    src = dst;
  }
}
